// GAT_67327907332631
// MI455X (gfx1250) — compile-verified, matrix operand order checked by reading, not run on hardware
//
#include <hip/hip_runtime.h>
#include <hip/hip_bf16.h>
#include <math.h>

// ---------------------------------------------------------------------------
// GAT attention forward, fused flash-style, f16 WMMA on gfx1250 (wave32).
//   query = ufea @ W^T + b
//   S     = leaky_relu2((query @ inter^T) / sqrt(128)), masked by adj
//   out   = softmax_row(S) @ inter
// adj (480 MB) is streamed exactly once -> non-temporal loads so it does not
// evict the L2-resident reused operands (inter f16 ~3MB, query f16 ~2.5MB).
// S is never materialized.
// ---------------------------------------------------------------------------

typedef __attribute__((ext_vector_type(16))) _Float16 v16h;
typedef __attribute__((ext_vector_type(8)))  _Float16 v8h;
typedef __attribute__((ext_vector_type(8)))  float    v8f;

#define DH      128           // hidden dim
#define BM      32            // user rows per workgroup (2 waves x 16)
#define BN      64            // item columns per tile
#define NWAVE   (BM / 16)
#define NEGINF  (-9e15f)
#define SCL     0.08838834764831845f   // 1/sqrt(128)

// Load one 16x16x32-f16 WMMA operand fragment for this lane.
// Documented 16-bit A/B layout: lane (lane&15) owns one row/column; lanes 0-15
// hold K = {kb+0..7, kb+16..23}, lanes 16-31 hold K = {kb+8..15, kb+24..31}.
// Caller passes p = row_base + kb + (lane>>4)*8 ; second chunk is +16 elems.
__device__ __forceinline__ v16h load_frag(const _Float16* p) {
  v8h lo = *(const v8h*)(p);
  v8h hi = *(const v8h*)(p + 16);
  return __builtin_shufflevector(lo, hi, 0, 1, 2, 3, 4, 5, 6, 7,
                                          8, 9, 10, 11, 12, 13, 14, 15);
}

// --------------------------- small prep kernels ----------------------------

__global__ void cvt_f16_kernel(const float* __restrict__ src,
                               _Float16* __restrict__ dst, int n) {
  int i = blockIdx.x * blockDim.x + threadIdx.x;
  if (i < n) dst[i] = (_Float16)src[i];
}

// query[row][col] = b[col] + dot(ufea[row,:], W[col,:])  (full f32, cast f16)
__global__ void query_kernel(const float* __restrict__ ufea,
                             const float* __restrict__ W,
                             const float* __restrict__ b,
                             _Float16* __restrict__ q16, int N) {
  int idx = blockIdx.x * blockDim.x + threadIdx.x;
  if (idx >= N * DH) return;
  int row = idx >> 7;
  int col = idx & (DH - 1);
  const float4* u = (const float4*)(ufea + (size_t)row * DH);
  const float4* w = (const float4*)(W + (size_t)col * DH);
  float acc = b[col];
#pragma unroll 8
  for (int k = 0; k < DH / 4; ++k) {
    float4 a = u[k], c = w[k];
    acc += a.x * c.x + a.y * c.y + a.z * c.z + a.w * c.w;
  }
  q16[idx] = (_Float16)acc;
}

// ------------------------- fused attention kernel --------------------------

__global__ __launch_bounds__(NWAVE * 32) void gat_attn_kernel(
    const _Float16* __restrict__ q16,   // [N,128] f16 query
    const _Float16* __restrict__ k16,   // [M,128] f16 inter (K and V)
    const int* __restrict__ adj,        // [N,M]
    float* __restrict__ out,            // [N,128]
    int N, int M) {
  __shared__ __align__(16) _Float16 Kv[BN * DH];          // item-major tile
  __shared__ __align__(16) _Float16 KvT[DH * BN];         // feature-major tile
  __shared__ __align__(16) _Float16 Ps[NWAVE][16 * BN];   // per-wave P tile

  const int tid  = threadIdx.x;
  const int wave = tid >> 5;
  const int lane = tid & 31;
  const int ln   = lane & 15;  // row (A) / column (B,C,D) owned by this lane
  const int hi   = lane >> 4;  // K-chunk half select / +8 row offset in C/D
  const int row0 = blockIdx.x * BM + wave * 16;

  // Q fragments for this wave's 16 rows: resident in VGPRs for the whole loop.
  v16h qf[4];
  {
    int r = row0 + ln;
    if (r > N - 1) r = N - 1;
    const _Float16* qp = q16 + (size_t)r * DH + hi * 8;
#pragma unroll
    for (int kk = 0; kk < 4; ++kk) qf[kk] = load_frag(qp + kk * 32);
  }

  v8f O[8];
#pragma unroll
  for (int t = 0; t < 8; ++t) O[t] = (v8f)0.0f;
  float mrow[8], lrow[8];
#pragma unroll
  for (int r = 0; r < 8; ++r) { mrow[r] = -INFINITY; lrow[r] = 0.0f; }

  // adj row this lane uses for prefetch hints
  int rowp = row0 + 8 * hi;
  if (rowp > N - 1) rowp = N - 1;

  const int ntile = (M + BN - 1) / BN;
  for (int jt = 0; jt < ntile; ++jt) {
    const int j0 = jt * BN;

    // Prefetch next adj tile (emits global_prefetch_b8).
    if (j0 + BN < M)
      __builtin_prefetch(adj + (size_t)rowp * M + j0 + BN, 0, 1);

    // Cooperative load of the inter tile into Kv (row-major) + KvT (transposed)
    for (int c = tid; c < BN * DH / 8; c += NWAVE * 32) {
      int item = c >> 4;            // 16 x 8-elem chunks per item row
      int fb   = (c & 15) * 8;
      int gi   = j0 + item;
      if (gi > M - 1) gi = M - 1;
      v8h v = *(const v8h*)(k16 + (size_t)gi * DH + fb);
      *(v8h*)(&Kv[item * DH + fb]) = v;
#pragma unroll
      for (int f = 0; f < 8; ++f) KvT[(fb + f) * BN + item] = v[f];
    }
    __syncthreads();

    // ---- S = Q @ K^T : 4 column tiles x 4 K-steps of v_wmma f16 ----
    v8f S[4];
#pragma unroll
    for (int ct = 0; ct < 4; ++ct) {
      v8f acc = (v8f)0.0f;
#pragma unroll
      for (int kk = 0; kk < 4; ++kk) {
        // B column n = item (ct*16+ln): its K-run = inter row, contiguous in Kv
        v16h bf = load_frag(&Kv[(ct * 16 + ln) * DH + kk * 32 + hi * 8]);
        acc = __builtin_amdgcn_wmma_f32_16x16x32_f16(
            false, qf[kk], false, bf, (short)0, acc, false, false);
      }
      S[ct] = acc;
    }

    // ---- scale, leaky_relu(slope 2), adj mask (D layout: row=r+8*hi, col=ln)
    //      adj is read exactly once -> non-temporal (TH=NT) keeps it out of L2
#pragma unroll
    for (int ct = 0; ct < 4; ++ct) {
      int col = j0 + ct * 16 + ln;
      bool cok = col < M;
#pragma unroll
      for (int r = 0; r < 8; ++r) {
        int row = row0 + r + 8 * hi;
        if (row > N - 1) row = N - 1;
        float s = S[ct][r] * SCL;
        s = s > 0.0f ? s : 2.0f * s;
        int a = cok ? __builtin_nontemporal_load(&adj[(size_t)row * M + col])
                    : 0;
        S[ct][r] = (a > 0) ? s : NEGINF;
      }
    }

    // ---- online softmax update ----
    float tm[8];
#pragma unroll
    for (int r = 0; r < 8; ++r)
      tm[r] = fmaxf(fmaxf(S[0][r], S[1][r]), fmaxf(S[2][r], S[3][r]));
#pragma unroll
    for (int msk = 1; msk < 16; msk <<= 1)    // stays inside 16-lane halves
#pragma unroll
      for (int r = 0; r < 8; ++r)
        tm[r] = fmaxf(tm[r], __shfl_xor(tm[r], msk, 32));

    float sc[8], mn[8], tl[8];
#pragma unroll
    for (int r = 0; r < 8; ++r) {
      mn[r] = fmaxf(mrow[r], tm[r]);          // finite: masked value is -9e15
      sc[r] = __expf(mrow[r] - mn[r]);        // first tile: exp(-inf)=0
      mrow[r] = mn[r];
      tl[r] = 0.0f;
    }
#pragma unroll
    for (int ct = 0; ct < 4; ++ct)
#pragma unroll
      for (int r = 0; r < 8; ++r) {
        float p = __expf(S[ct][r] - mn[r]);
        tl[r] += p;
        Ps[wave][(r + 8 * hi) * BN + ct * 16 + ln] = (_Float16)p;
      }
#pragma unroll
    for (int msk = 1; msk < 16; msk <<= 1)
#pragma unroll
      for (int r = 0; r < 8; ++r) tl[r] += __shfl_xor(tl[r], msk, 32);
#pragma unroll
    for (int r = 0; r < 8; ++r) lrow[r] = lrow[r] * sc[r] + tl[r];
#pragma unroll
    for (int t = 0; t < 8; ++t)
#pragma unroll
      for (int r = 0; r < 8; ++r) O[t][r] *= sc[r];

    // ---- O += P @ V : 2 K-steps x 8 feature tiles of v_wmma f16 ----
    // (Ps is wave-private; same-wave DS ordering + compiler waits cover RAW)
#pragma unroll
    for (int kk = 0; kk < 2; ++kk) {
      v16h ap = load_frag(&Ps[wave][ln * BN + kk * 32 + hi * 8]);
#pragma unroll
      for (int ct = 0; ct < 8; ++ct) {
        // B column n = feature (ct*16+ln): its K-run over items is contiguous
        // in the transposed tile KvT.
        v16h bv = load_frag(&KvT[(ct * 16 + ln) * BN + kk * 32 + hi * 8]);
        O[ct] = __builtin_amdgcn_wmma_f32_16x16x32_f16(
            false, ap, false, bv, (short)0, O[ct], false, false);
      }
    }
    __syncthreads();  // before next tile overwrites Kv/KvT
  }

  // ---- epilogue: out = O / l  (write-once -> non-temporal stores) ----
  float inv[8];
#pragma unroll
  for (int r = 0; r < 8; ++r) inv[r] = 1.0f / lrow[r];
#pragma unroll
  for (int ct = 0; ct < 8; ++ct)
#pragma unroll
    for (int r = 0; r < 8; ++r) {
      int row = row0 + r + 8 * hi;
      if (row < N)
        __builtin_nontemporal_store(O[ct][r] * inv[r],
                                    &out[(size_t)row * DH + ct * 16 + ln]);
    }
}

// ------------------------------- launcher ----------------------------------

extern "C" void kernel_launch(void* const* d_in, const int* in_sizes, int n_in,
                              void* d_out, int out_size, void* d_ws,
                              size_t ws_size, hipStream_t stream) {
  const float* ufea  = (const float*)d_in[0];  // [N,128]
  const float* inter = (const float*)d_in[1];  // [M,128]
  const int*   adj   = (const int*)d_in[2];    // [N,M]
  const float* W     = (const float*)d_in[3];  // [128,128]
  const float* b     = (const float*)d_in[4];  // [128]
  float* out = (float*)d_out;

  const int N = in_sizes[0] / DH;  // 10000
  const int M = in_sizes[1] / DH;  // 12000

  // workspace: q16 [N*128] f16, k16 [M*128] f16  (~5.6 MB total)
  _Float16* q16 = (_Float16*)d_ws;
  _Float16* k16 = q16 + (size_t)N * DH;

  cvt_f16_kernel<<<(M * DH + 255) / 256, 256, 0, stream>>>(inter, k16, M * DH);
  query_kernel<<<(N * DH + 255) / 256, 256, 0, stream>>>(ufea, W, b, q16, N);

  dim3 grid((N + BM - 1) / BM);                  // 313 workgroups
  gat_attn_kernel<<<grid, NWAVE * 32, 0, stream>>>(q16, k16, adj, out, N, M);
}
